// JumpAwareHead_66340064854215
// MI455X (gfx1250) — matrix operand order repaired and checked by reading, not compiled, not run on hardware
//
#include <hip/hip_runtime.h>
#include <hip/hip_bf16.h>

// Problem constants (from reference)
#define BB 16       // batch
#define LL 1024     // seq len
#define DA 7        // activation dim
#define DI 512      // d_inner
#define DS 16       // d_state
#define TT (LL + 1) // 1025 timesteps after concat
#define RR (BB * TT) // 16400 total rows; 16400/16 = 1025 exact M-tiles

#define MTILES 1025          // 16-row M tiles
#define MPAIRS ((MTILES + 1) / 2)   // 513 (last pair half-padded)
#define NTILES 32            // 16-col N tiles
#define NGROUPS (NTILES / 4) // 8 groups of 4 N-tiles

typedef __attribute__((ext_vector_type(16))) __bf16 v16bf;
typedef __attribute__((ext_vector_type(8)))  float  v8f;

__device__ __forceinline__ unsigned short f32_to_bf16_rne(float f) {
    unsigned int u = __float_as_uint(f);
    u += 0x7FFFu + ((u >> 16) & 1u);   // round-to-nearest-even
    return (unsigned short)(u >> 16);
}

// ---------------------------------------------------------------------------
// Kernel 1: per-row x = seq @ We.T + be (K=7, scalar), store bf16 copy for the
// WMMA GEMM, and compute Bm/Cm rows (exact f32) from the LDS-resident x row.
// grid = RR blocks, block = 512 threads.
// ---------------------------------------------------------------------------
__global__ void prep_kernel(const float* __restrict__ A_act,
                            const float* __restrict__ a_new,
                            const float* __restrict__ We,
                            const float* __restrict__ be,
                            const float* __restrict__ WB,
                            const float* __restrict__ WC,
                            unsigned short* __restrict__ Xb,
                            float* __restrict__ Bm,
                            float* __restrict__ Cm) {
    __shared__ float xrow[DI];
    int r = blockIdx.x;
    int b = r / TT;
    int t = r % TT;
    const float* srow = (t < LL) ? (A_act + ((size_t)b * LL + t) * DA)
                                 : (a_new + (size_t)b * DA);
    float sv[DA];
#pragma unroll
    for (int k = 0; k < DA; ++k) sv[k] = srow[k];   // wave-broadcast loads

    int n = threadIdx.x;
    float xv = be[n];
#pragma unroll
    for (int k = 0; k < DA; ++k) xv = fmaf(We[n * DA + k], sv[k], xv);
    xrow[n] = xv;
    Xb[(size_t)r * DI + n] = f32_to_bf16_rne(xv);
    __syncthreads();

    // Bm[r,s] / Cm[r,s] = <x_row, WB/WC row s>
    if (threadIdx.x < 2 * DS) {
        int sidx = threadIdx.x & (DS - 1);
        const float* Wrow = (threadIdx.x < DS ? WB : WC) + (size_t)sidx * DI;
        float acc = 0.0f;
        for (int i = 0; i < DI; ++i) acc = fmaf(xrow[i], Wrow[i], acc);
        if (threadIdx.x < DS) Bm[(size_t)r * DS + sidx] = acc;
        else                  Cm[(size_t)r * DS + sidx] = acc;
    }
}

// ---------------------------------------------------------------------------
// Kernel 2: Wd f32 -> bf16
// ---------------------------------------------------------------------------
__global__ void cvt_wd_kernel(const float* __restrict__ Wd,
                              unsigned short* __restrict__ Wdb) {
    int i = blockIdx.x * blockDim.x + threadIdx.x;
    if (i < DI * DI) Wdb[i] = f32_to_bf16_rne(Wd[i]);
}

// ---------------------------------------------------------------------------
// Kernel 3: delta = softplus(X @ Wd.T + bd) via v_wmma_f32_16x16x32_bf16.
// Register-tiled: each wave computes a 2(M) x 4(N) block of 16x16 tiles
// (8 accumulators), reusing each A fragment 4x and each B fragment 2x.
//   L2 traffic: ~1.07 GB (1x1 tiles) -> ~0.40 GB.
// Fragment layouts per CDNA5 ISA 7.12.2:
//   A 16x32 bf16 : lane l<16 -> M=l ; K half = (lane/16)*8 ; elems 0-7 -> K+0..7,
//                  elems 8-15 -> K+16..23
//   B 32x16 bf16 : lane -> N=lane%16 ; K half = (lane/16)*8 ; elems 0-7 -> K+0..7, 8-15 -> K+16..23
//   C 16x16 f32  : lane -> N=lane%16 ; vgpr j -> M = j + 8*(lane/16)
// M padded to 1026 tiles (513 pairs): padding A-loads land inside the adjacent
// Wdb workspace region (valid memory, finite bf16); padding stores are guarded.
// ---------------------------------------------------------------------------
__global__ void gemm_delta_kernel(const unsigned short* __restrict__ Xb,
                                  const unsigned short* __restrict__ Wdb,
                                  const float* __restrict__ bd,
                                  float* __restrict__ delta) {
    int wave = (int)((blockIdx.x * blockDim.x + threadIdx.x) >> 5);
    int lane = threadIdx.x & 31;
    int mp = wave >> 3;          // M-pair index, 0..512
    int ng = wave & 7;           // N-group index, 0..7
    if (mp >= MPAIRS) return;

    int half = lane >> 4;
    int l    = lane & 15;

    const unsigned short* arow0 = Xb + ((size_t)((mp * 2 + 0) * 16 + l)) * DI + half * 8;
    const unsigned short* arow1 = Xb + ((size_t)((mp * 2 + 1) * 16 + l)) * DI + half * 8;
    const unsigned short* brow[4];
#pragma unroll
    for (int ni = 0; ni < 4; ++ni)
        brow[ni] = Wdb + ((size_t)((ng * 4 + ni) * 16 + l)) * DI + half * 8;

    v8f acc[2][4];
#pragma unroll
    for (int mi = 0; mi < 2; ++mi)
#pragma unroll
        for (int ni = 0; ni < 4; ++ni)
            acc[mi][ni] = (v8f){0.f, 0.f, 0.f, 0.f, 0.f, 0.f, 0.f, 0.f};

    for (int kc = 0; kc < DI; kc += 32) {
        union { unsigned short us[16]; v16bf v; } af0, af1, bf[4];
#pragma unroll
        for (int e = 0; e < 8; ++e) {
            af0.us[e]     = arow0[kc + e];        // K = kc + half*8 + e
            af0.us[8 + e] = arow0[kc + 16 + e];   // K = kc + 16 + half*8 + e
            af1.us[e]     = arow1[kc + e];
            af1.us[8 + e] = arow1[kc + 16 + e];
        }
#pragma unroll
        for (int ni = 0; ni < 4; ++ni)
#pragma unroll
            for (int e = 0; e < 16; ++e) bf[ni].us[e] = brow[ni][kc + (e < 8 ? e : 8 + e)]; // K = kc+half*8+e (e<8), kc+16+half*8+(e-8)

#pragma unroll
        for (int ni = 0; ni < 4; ++ni) {
            acc[0][ni] = __builtin_amdgcn_wmma_f32_16x16x32_bf16(
                false, af0.v, false, bf[ni].v, (short)0, acc[0][ni], false, false);
            acc[1][ni] = __builtin_amdgcn_wmma_f32_16x16x32_bf16(
                false, af1.v, false, bf[ni].v, (short)0, acc[1][ni], false, false);
        }
    }

#pragma unroll
    for (int ni = 0; ni < 4; ++ni) {
        int n = (ng * 4 + ni) * 16 + l;
        float bdn = bd[n];
#pragma unroll
        for (int mi = 0; mi < 2; ++mi) {
            int mbase = (mp * 2 + mi) * 16 + half * 8;
#pragma unroll
            for (int j = 0; j < 8; ++j) {
                int m = mbase + j;
                if (m < RR) {
                    float z = acc[mi][ni][j] + bdn;
                    float sp = (z > 20.0f) ? z : __logf(1.0f + __expf(z)); // softplus
                    delta[(size_t)m * DI + n] = sp;
                }
            }
        }
    }
}

// ---------------------------------------------------------------------------
// Kernel 4: sequential selective scan. One lane per (b, d) chain; 16 states in
// registers. x_t recomputed from seq (K=7 broadcast FMAs) in exact f32.
// grid = BB*4 blocks, block = 128 threads (d = 128 per block).
// ---------------------------------------------------------------------------
__global__ void scan_kernel(const float* __restrict__ A_act,
                            const float* __restrict__ a_new,
                            const float* __restrict__ We,
                            const float* __restrict__ be,
                            const float* __restrict__ A_log,
                            const float* __restrict__ Dvec,
                            const float* __restrict__ delta,
                            const float* __restrict__ Bm,
                            const float* __restrict__ Cm,
                            float* __restrict__ yb) {
    int b = blockIdx.x >> 2;
    int d = ((blockIdx.x & 3) << 7) + threadIdx.x;

    float Wer[DA];
#pragma unroll
    for (int k = 0; k < DA; ++k) Wer[k] = We[d * DA + k];
    float bee = be[d];

    float Ac[DS];
#pragma unroll
    for (int s = 0; s < DS; ++s) Ac[s] = -__expf(A_log[d * DS + s]); // A = -exp(A_log)

    float h[DS];
#pragma unroll
    for (int s = 0; s < DS; ++s) h[s] = 0.0f;

    float x_t = 0.0f;
    for (int t = 0; t < TT; ++t) {
        size_t r = (size_t)b * TT + t;
        const float* srow = (t < LL) ? (A_act + ((size_t)b * LL + t) * DA)
                                     : (a_new + (size_t)b * DA);
        x_t = bee;
#pragma unroll
        for (int k = 0; k < DA; ++k) x_t = fmaf(Wer[k], srow[k], x_t);

        float dv  = delta[r * DI + d];       // coalesced across lanes
        float dvx = dv * x_t;
#pragma unroll
        for (int s = 0; s < DS; ++s) {
            float bm = Bm[r * DS + s];       // wave-broadcast
            // h = exp(delta*A)*h + delta*B*x
            h[s] = fmaf(__expf(dv * Ac[s]), h[s], dvx * bm);
        }
    }

    // y at the last timestep only (all the reference uses)
    size_t rl = (size_t)b * TT + LL;
    float y = Dvec[d] * x_t;
#pragma unroll
    for (int s = 0; s < DS; ++s) y = fmaf(h[s], Cm[rl * DS + s], y);
    yb[(size_t)b * DI + d] = y;
}

// ---------------------------------------------------------------------------
// Kernel 5: LayerNorm over d, logit = <out, Wc> + bc, sigmoid.
// grid = BB blocks, block = 512 threads.
// ---------------------------------------------------------------------------
__global__ void finalize_kernel(const float* __restrict__ yb,
                                const float* __restrict__ ln_g,
                                const float* __restrict__ ln_b,
                                const float* __restrict__ Wc,
                                const float* __restrict__ bc,
                                float* __restrict__ out) {
    __shared__ float red[DI];
    int b = blockIdx.x;
    int d = threadIdx.x;
    float v = yb[(size_t)b * DI + d];

    red[d] = v;
    __syncthreads();
    for (int off = DI / 2; off > 0; off >>= 1) {
        if (d < off) red[d] += red[d + off];
        __syncthreads();
    }
    float mu = red[0] * (1.0f / DI);
    __syncthreads();

    float c = v - mu;
    red[d] = c * c;
    __syncthreads();
    for (int off = DI / 2; off > 0; off >>= 1) {
        if (d < off) red[d] += red[d + off];
        __syncthreads();
    }
    float var = red[0] * (1.0f / DI);
    __syncthreads();

    float o = c * rsqrtf(var + 1e-5f) * ln_g[d] + ln_b[d];
    red[d] = o * Wc[d];
    __syncthreads();
    for (int off = DI / 2; off > 0; off >>= 1) {
        if (d < off) red[d] += red[d + off];
        __syncthreads();
    }
    if (d == 0) {
        float logit = red[0] + bc[0];
        out[b]      = 1.0f / (1.0f + __expf(-logit));  // sigmoid
        out[BB + b] = logit;
    }
}

// ---------------------------------------------------------------------------
// Host launcher
// ---------------------------------------------------------------------------
extern "C" void kernel_launch(void* const* d_in, const int* in_sizes, int n_in,
                              void* d_out, int out_size, void* d_ws, size_t ws_size,
                              hipStream_t stream) {
    const float* A_act = (const float*)d_in[0];   // (B, L, DA)
    const float* a_new = (const float*)d_in[1];   // (B, DA)
    const float* We    = (const float*)d_in[2];   // (DI, DA)
    const float* be    = (const float*)d_in[3];   // (DI)
    const float* Wd    = (const float*)d_in[4];   // (DI, DI)
    const float* bd    = (const float*)d_in[5];   // (DI)
    const float* WB    = (const float*)d_in[6];   // (DS, DI)
    const float* WC    = (const float*)d_in[7];   // (DS, DI)
    const float* Dvec  = (const float*)d_in[8];   // (DI)
    const float* A_log = (const float*)d_in[9];   // (DI, DS)
    const float* ln_g  = (const float*)d_in[10];  // (DI)
    const float* ln_b  = (const float*)d_in[11];  // (DI)
    const float* Wc    = (const float*)d_in[12];  // (1, DI)
    const float* bc    = (const float*)d_in[13];  // (1)
    float* out = (float*)d_out;                   // [16 sigmoid | 16 logit]

    // Workspace layout (256B-aligned sections), ~53 MB total
    const size_t XB_BYTES    = (size_t)RR * DI * sizeof(unsigned short); // 16,793,600
    const size_t WDB_BYTES   = (size_t)DI * DI * sizeof(unsigned short); //    524,288
    const size_t DELTA_BYTES = (size_t)RR * DI * sizeof(float);          // 33,587,200
    const size_t BM_BYTES    = (size_t)RR * DS * sizeof(float);          //  1,049,600

    char* ws = (char*)d_ws;
    unsigned short* Xb    = (unsigned short*)(ws);
    unsigned short* Wdb   = (unsigned short*)(ws + XB_BYTES);
    float*          delta = (float*)(ws + XB_BYTES + WDB_BYTES);
    float*          Bm    = (float*)(ws + XB_BYTES + WDB_BYTES + DELTA_BYTES);
    float*          Cm    = (float*)(ws + XB_BYTES + WDB_BYTES + DELTA_BYTES + BM_BYTES);
    float*          yb    = (float*)(ws + XB_BYTES + WDB_BYTES + DELTA_BYTES + 2 * BM_BYTES);

    // 1) x rows (f32->bf16) + Bm/Cm rows
    prep_kernel<<<RR, DI, 0, stream>>>(A_act, a_new, We, be, WB, WC, Xb, Bm, Cm);

    // 2) Wd -> bf16
    cvt_wd_kernel<<<(DI * DI + 255) / 256, 256, 0, stream>>>(Wd, Wdb);

    // 3) WMMA GEMM + softplus epilogue: 513 M-pairs x 8 N-groups waves,
    //    8 waves/block -> 513 blocks of 256 threads
    gemm_delta_kernel<<<(MPAIRS * NGROUPS) / 8, 256, 0, stream>>>(Xb, Wdb, bd, delta);

    // 4) selective scan: 8192 lanes, 1025 steps each
    scan_kernel<<<BB * 4, 128, 0, stream>>>(A_act, a_new, We, be, A_log, Dvec,
                                            delta, Bm, Cm, yb);

    // 5) LayerNorm + logit + sigmoid
    finalize_kernel<<<BB, DI, 0, stream>>>(yb, ln_g, ln_b, Wc, bc, out);
}
